// LogGNN_90091234001458
// MI455X (gfx1250) — matrix operand order repaired and checked by reading, not compiled, not run on hardware
//
#include <hip/hip_runtime.h>

typedef __attribute__((ext_vector_type(16))) __bf16  v16bf;
typedef __attribute__((ext_vector_type(8)))  float   v8f;
typedef __attribute__((ext_vector_type(4)))  unsigned int v4u;
typedef __attribute__((ext_vector_type(4)))  unsigned int u32x4;
typedef __attribute__((ext_vector_type(8)))  int     i32x8;
typedef __attribute__((ext_vector_type(4)))  int     i32x4;

union Frag { v16bf v; v4u q[2]; };

#if defined(__has_builtin)
# if __has_builtin(__builtin_amdgcn_tensor_load_to_lds) && __has_builtin(__builtin_amdgcn_s_wait_tensorcnt)
#  define USE_TDM 1
# endif
#endif

__device__ __forceinline__ unsigned short f2bf(float f) {
    unsigned u = __float_as_uint(f);
    unsigned r = u + 0x7FFFu + ((u >> 16) & 1u);   // round-to-nearest-even
    return (unsigned short)(r >> 16);
}

// Order-preserving float->uint encoding for atomicMax-based segment max.
__device__ __forceinline__ unsigned enc_f32(float f) {
    unsigned u = __float_as_uint(f);
    return (u & 0x80000000u) ? ~u : (u | 0x80000000u);
}
__device__ __forceinline__ float dec_f32(unsigned e) {
    unsigned u = (e & 0x80000000u) ? (e & 0x7FFFFFFFu) : ~e;
    return __uint_as_float(u);
}

// ---------------- elementwise converters ----------------
__global__ void k_conv_bf(const float* __restrict__ in, unsigned short* __restrict__ out, int n) {
    int t = blockIdx.x * blockDim.x + threadIdx.x;
    if (t < n) out[t] = f2bf(in[t]);
}

// W[k][n] (128x128 row-major) -> Wt[n][k] bf16
__global__ void k_tconv(const float* __restrict__ W, unsigned short* __restrict__ Wt) {
    int t = blockIdx.x * blockDim.x + threadIdx.x;
    if (t < 128 * 128) {
        int n = t >> 7, k = t & 127;
        Wt[t] = f2bf(W[k * 128 + n]);
    }
}

__global__ void k_zero4(float4* __restrict__ p, int n4) {
    int t = blockIdx.x * blockDim.x + threadIdx.x;
    if (t < n4) p[t] = make_float4(0.f, 0.f, 0.f, 0.f);
}

// ---------------- WMMA GEMM: C[M,128] = Abf[M,128] x W (Wt = W^T bf16 [128][128]) ----------------
// block = 256 threads = 8 waves; wave w owns column tile [w*16,w*16+16); block strip = 16 rows.
// A strip (16x128 bf16 = 4KB) is staged once per block into LDS via the Tensor Data Mover,
// with a 16B pad per 256B row -> LDS row stride 272B = 68 dwords (4 banks mod 64):
// the 16-row ds_load_b128 A-fragment reads are bank-conflict-free.
#define LDS_ROW 136   // elements per padded LDS row (128 data + 8 pad)

template <bool HAS_BIAS, bool BF16_OUT>
__global__ __launch_bounds__(256)
void k_gemm_bf16(const unsigned short* __restrict__ Abf,
                 const unsigned short* __restrict__ Wt,
                 const float* __restrict__ bias,
                 float* __restrict__ C,
                 unsigned short* __restrict__ Cbf) {
    __shared__ __align__(16) unsigned short shA[16 * LDS_ROW + 8];

    const int wave    = threadIdx.x >> 5;
    const int lane    = threadIdx.x & 31;
    const int tileRow = blockIdx.x;
    const int colBase = wave * 16;
    const int nn      = lane & 15;
    const int grp     = lane >> 4;

#ifdef USE_TDM
    if (threadIdx.x < 32) {
        // Build Tensor DMA Descriptor (D#) per CDNA5 ISA ch.8: 2D tile 16x128, 2-byte elems.
        unsigned long long ga =
            (unsigned long long)(const void*)(Abf + (size_t)tileRow * 16 * 128);
        unsigned lds_off = (unsigned)(unsigned long long)(const void*)&shA[0];
        u32x4 g0;
        g0.x = 1u;                                           // count=1, gather off
        g0.y = lds_off;                                      // lds_addr (bytes)
        g0.z = (unsigned)(ga & 0xFFFFFFFFu);                 // global_addr[31:0]
        g0.w = (unsigned)((ga >> 32) & 0x01FFFFFFu) | (2u << 30);  // addr[56:32], type=2
        i32x8 g1;
        g1[0] = (1 << 16)            // data_size = 2B
              | (1 << 20)            // pad_enable
              | (5 << 22)            // pad_interval: 64 dwords (256B) per row
              | (3 << 25);           // pad_amount: 4 dwords (16B)
        g1[1] = (128 << 16);         // tensor_dim0 = 128 elems (bits 79:48)
        g1[2] = (16 << 16);          // tensor_dim1 = 16 rows (bits 111:80)
        g1[3] = (128 << 16);         // tile_dim0 = 128 (bits 127:112)
        g1[4] = 16;                  // tile_dim1 = 16, tile_dim2 = 0
        g1[5] = 128;                 // tensor_dim0_stride = 128 elems
        g1[6] = 0;                   // stride hi / dim1_stride lo
        g1[7] = 0;
        i32x4 z4 = {0, 0, 0, 0};                 // groups 2/3 unused (2D tensor)
        i32x8 z8 = {0, 0, 0, 0, 0, 0, 0, 0};     // 6-arg toolchain extra group
        __builtin_amdgcn_tensor_load_to_lds(g0, g1, z4, z4, z8, 0);
        __builtin_amdgcn_s_wait_tensorcnt(0);
    }
#else
    {   // cooperative fallback: 256 threads x 16B = 4KB strip
        int r = threadIdx.x >> 4, cc = threadIdx.x & 15;
        *(v4u*)&shA[r * LDS_ROW + cc * 8] =
            *(const v4u*)(Abf + (size_t)(tileRow * 16 + r) * 128 + cc * 8);
    }
#endif
    __syncthreads();

    const unsigned short* Arow = shA + nn * LDS_ROW;
    const unsigned short* Bcol = Wt + (size_t)(colBase + nn) * 128 + 16 * grp;

    v8f c = {0.f, 0.f, 0.f, 0.f, 0.f, 0.f, 0.f, 0.f};
#pragma unroll
    for (int kt = 0; kt < 128; kt += 32) {
        Frag a, b;
        // A 16x32 bf16 layout: lanes<16 -> K=kt..kt+15; lanes>=16 -> K=kt+16..kt+31
        a.q[0] = *(const v4u*)(Arow + kt + 16 * grp);
        a.q[1] = *(const v4u*)(Arow + kt + 8 + 16 * grp);
        // B 32x16 bf16 layout: N = lane%16; lanes<16 -> K=kt..kt+15; lanes>=16 -> K=kt+16..kt+31
        b.q[0] = *(const v4u*)(Bcol + kt);
        b.q[1] = *(const v4u*)(Bcol + kt + 8);
        c = __builtin_amdgcn_wmma_f32_16x16x32_bf16(false, a.v, false, b.v,
                                                    (short)0, c, false, false);
    }

    float bb = 0.f;
    if (HAS_BIAS) bb = bias[colBase + nn];
#pragma unroll
    for (int v = 0; v < 8; ++v) {                       // C/D: lanes<16 rows 0..7, lanes>=16 rows 8..15
        int grow = tileRow * 16 + v + 8 * grp;          // M is always a multiple of 16 here
        float val = c[v] + bb;
        size_t off = (size_t)grow * 128 + colBase + nn;
        C[off] = val;
        if (BF16_OUT) Cbf[off] = f2bf(val);
    }
}

// ---------------- s/d attention dot products: wave per node ----------------
__global__ void k_sd(const float* __restrict__ g, const float* __restrict__ as,
                     const float* __restrict__ ad, float* __restrict__ sv,
                     float* __restrict__ dv, int Nn) {
    int wave = (int)((blockIdx.x * blockDim.x + threadIdx.x) >> 5);
    int lane = threadIdx.x & 31;
    if (wave >= Nn) return;
    float4 gv = *(const float4*)(g + (size_t)wave * 128 + lane * 4);
    float4 a1 = *(const float4*)(as + lane * 4);
    float4 a2 = *(const float4*)(ad + lane * 4);
    float ps = gv.x * a1.x + gv.y * a1.y + gv.z * a1.z + gv.w * a1.w;
    float pd = gv.x * a2.x + gv.y * a2.y + gv.z * a2.z + gv.w * a2.w;
#pragma unroll
    for (int off = 16; off > 0; off >>= 1) {
        ps += __shfl_xor(ps, off, 32);
        pd += __shfl_xor(pd, off, 32);
    }
    if (lane == 0) { sv[wave] = ps; dv[wave] = pd; }
}

// ---------------- edge pass 1: logits + segment max (self-loops at t>=E) ----------------
__global__ void k_logits(const float* __restrict__ sv, const float* __restrict__ dv,
                         const int* __restrict__ src, const int* __restrict__ dst,
                         float* __restrict__ lg, unsigned* __restrict__ menc,
                         int E_, int ET_) {
    int t = blockIdx.x * blockDim.x + threadIdx.x;
    if (t >= ET_) return;
    int s_ = (t < E_) ? src[t] : (t - E_);
    int d_ = (t < E_) ? dst[t] : (t - E_);
    float e = sv[s_] + dv[d_];
    e = (e > 0.f) ? e : 0.2f * e;                          // leaky_relu, slope 0.2
    lg[t] = e;
    atomicMax(menc + d_, enc_f32(e));
}

// ---------------- edge pass 2: exp + z accumulation + weighted gather/scatter ----------------
// wave per edge; lane0 handles the softmax denominator, all lanes scatter 4 features each.
__global__ void k_scatter(const float* __restrict__ g, const float* __restrict__ lg,
                          const unsigned* __restrict__ menc,
                          const int* __restrict__ src, const int* __restrict__ dst,
                          float* __restrict__ accum, float* __restrict__ z,
                          int E_, int ET_) {
    int wave = (int)((blockIdx.x * blockDim.x + threadIdx.x) >> 5);
    int lane = threadIdx.x & 31;
    if (wave >= ET_) return;
    int s_ = (wave < E_) ? src[wave] : (wave - E_);
    int d_ = (wave < E_) ? dst[wave] : (wave - E_);
    float ex = __expf(lg[wave] - dec_f32(menc[d_]));       // m finite: self-loops guarantee it
    if (lane == 0) atomicAdd(z + d_, ex);
    float4 v = *(const float4*)(g + (size_t)s_ * 128 + lane * 4);
    float* ap = accum + (size_t)d_ * 128 + lane * 4;
    atomicAdd(ap + 0, v.x * ex);
    atomicAdd(ap + 1, v.y * ex);
    atomicAdd(ap + 2, v.z * ex);
    atomicAdd(ap + 3, v.w * ex);
}

// ---------------- finalize: h = accum / z + bias (+ bf16 copy for next GEMM) ----------------
template <bool BF16_OUT>
__global__ void k_finalize(const float* __restrict__ accum, const float* __restrict__ z,
                           const float* __restrict__ bias, float* __restrict__ hout,
                           unsigned short* __restrict__ hbf, int total) {
    int t = blockIdx.x * blockDim.x + threadIdx.x;
    if (t >= total) return;
    int n = t >> 7, f = t & 127;
    float v = accum[t] / z[n] + bias[f];
    hout[t] = v;
    if (BF16_OUT) hbf[t] = f2bf(v);
}

extern "C" void kernel_launch(void* const* d_in, const int* in_sizes, int n_in,
                              void* d_out, int out_size, void* d_ws, size_t ws_size,
                              hipStream_t stream) {
    const float* x        = (const float*)d_in[0];
    const int*   ei       = (const int*)d_in[1];
    const float* W_node   = (const float*)d_in[3];
    const float* b_node   = (const float*)d_in[4];
    const float* lyr_W    = (const float*)d_in[9];
    const float* att_src  = (const float*)d_in[10];
    const float* att_dst  = (const float*)d_in[11];
    const float* lyr_bias = (const float*)d_in[12];
    float* out = (float*)d_out;

    const int N  = in_sizes[0] / 128;      // 50000
    const int E  = in_sizes[1] / 2;        // 800000
    const int ET = E + N;                  // + self loops
    const int* src = ei;
    const int* dst = ei + E;

    // ---- workspace layout (256B aligned slabs) ----
    char* p = (char*)d_ws;
    auto alloc = [&](size_t bytes) -> char* {
        char* r = p;
        p += (bytes + 255) & ~(size_t)255;
        return r;
    };
    unsigned short* xbf   = (unsigned short*)alloc((size_t)N * 128 * 2);
    unsigned short* hbf   = (unsigned short*)alloc((size_t)N * 128 * 2);
    float*          h     = (float*)alloc((size_t)N * 128 * 4);
    float*          g     = (float*)alloc((size_t)N * 128 * 4);
    float*          accum = (float*)alloc((size_t)N * 130 * 4);   // accum | z | menc (zeroed together)
    float*          z     = accum + (size_t)N * 128;
    unsigned*       menc  = (unsigned*)(z + N);
    float*          sv    = (float*)alloc((size_t)N * 4);
    float*          dv    = (float*)alloc((size_t)N * 4);
    float*          lg    = (float*)alloc((size_t)ET * 4);
    unsigned short* WtN   = (unsigned short*)alloc(128 * 128 * 2);
    unsigned short* Wt0   = (unsigned short*)alloc(128 * 128 * 2);
    unsigned short* Wt1   = (unsigned short*)alloc(128 * 128 * 2);

    const int B = 256;
    const int nNF   = N * 128;
    const int gConv = (nNF + B - 1) / B;
    const int gW    = (128 * 128 + B - 1) / B;
    const int gGemm = N / 16;                                // 3125, exact
    const int gSd   = (N + 7) / 8;                           // wave per node
    const int nZ4   = (N * 130) / 4;
    const int gZero = (nZ4 + B - 1) / B;
    const int gEdge = (ET + B - 1) / B;
    const int gScat = (ET + 7) / 8;                          // wave per edge
    const int gFin  = (nNF + B - 1) / B;

    // encoders / weight prep (edge-encoder outputs are dead code in the reference -> skipped)
    k_conv_bf<<<gConv, B, 0, stream>>>(x, xbf, nNF);
    k_tconv<<<gW, B, 0, stream>>>(W_node, WtN);
    k_tconv<<<gW, B, 0, stream>>>(lyr_W, Wt0);
    k_tconv<<<gW, B, 0, stream>>>(lyr_W + 128 * 128, Wt1);

    // h = x @ W_node + b_node  (fused bias + bf16 copy)
    k_gemm_bf16<true, true><<<gGemm, B, 0, stream>>>(xbf, WtN, b_node, h, hbf);

    for (int l = 0; l < 2; ++l) {
        const unsigned short* Wt = (l == 0) ? Wt0 : Wt1;
        // g = h @ W[l]
        k_gemm_bf16<false, false><<<gGemm, B, 0, stream>>>(hbf, Wt, nullptr, g, nullptr);
        // per-node attention contributions
        k_sd<<<gSd, B, 0, stream>>>(g, att_src + l * 128, att_dst + l * 128, sv, dv, N);
        // clear accum | z | menc
        k_zero4<<<gZero, B, 0, stream>>>((float4*)accum, nZ4);
        // segment softmax over dst (self-loops included) + weighted aggregation
        k_logits<<<gEdge, B, 0, stream>>>(sv, dv, src, dst, lg, menc, E, ET);
        k_scatter<<<gScat, B, 0, stream>>>(g, lg, menc, src, dst, accum, z, E, ET);
        // h = accum / z + bias  (last layer -> d_out, no bf16 copy needed)
        if (l == 1) {
            k_finalize<false><<<gFin, B, 0, stream>>>(accum, z, lyr_bias + l * 128, out, nullptr, nNF);
        } else {
            k_finalize<true><<<gFin, B, 0, stream>>>(accum, z, lyr_bias + l * 128, h, hbf, nNF);
        }
    }
}
